// FlatLadder7x7NNUEv1_11458972745779
// MI455X (gfx1250) — matrix operand order repaired and checked by reading, not compiled, not run on hardware
//
#include <hip/hip_runtime.h>
#include <hip/hip_bf16.h>

typedef _Float16 h16;
typedef __attribute__((ext_vector_type(16))) _Float16 v16h;
typedef __attribute__((ext_vector_type(8)))  _Float16 v8h;
typedef __attribute__((ext_vector_type(8)))  float    v8f;

#define DMC   128
#define NBAT  16384
#define QMAXF (127.0f/128.0f)

// f16 weight image offsets inside d_ws (in halves)
#define LW23_OFF 0        // 2*3*128*128 = 98304
#define CW_OFF   98304    // 3*128*128   = 49152
#define CWF_OFF  147456   // 48*128      = 6144
#define WF_TOTAL 153600

// Defeat LICM/CSE on an index so weight/bias loads stay inside their stage
// (keeps VGPR pressure bounded; the loads are L0/L2-resident anyway).
#define OPQ(v_) asm volatile("" : "+v"(v_))

// Branchless mish: tanh(softplus(x)) = (s^2-1)/(s^2+1), s = 1+e^x
// => mish(x) = x * n / (n + 2),  n = t*(t+2), t = e^x.
__device__ __forceinline__ float mishf(float x) {
    float t = __expf(fminf(x, 18.0f));     // e^18^2 ~ 4e15, finite
    float n = t * (t + 2.0f);
    return x * n * __builtin_amdgcn_rcpf(n + 2.0f);
}

__device__ __forceinline__ v8f wmma_f16(v16h a, v16h b, v8f c) {
    return __builtin_amdgcn_wmma_f32_16x16x32_f16(false, a, false, b, (short)0, c, false, false);
}

// A fragment: weights, global row-major [out][128] f16. 16x32 tile at (rowBase, 32*ks).
__device__ __forceinline__ v16h load_a_frag(const h16* W, int rowBase, int ks, int lane) {
    int off = (rowBase + (lane & 15)) * DMC + ks * 32 + ((lane >> 4) << 3);
    OPQ(off);                               // re-load per use, don't hoist
    const h16* p = W + off;
    v8h lo = *(const v8h*)p;                // K base .. base+7
    v8h hi = *(const v8h*)(p + 16);         // K base+16 .. base+23
    v16h r;
#pragma unroll
    for (int i = 0; i < 8; ++i) { r[i] = lo[i]; r[i + 8] = hi[i]; }
    return r;
}

// B fragment: activations in LDS, [col][128] f16. 32x16 tile: cols colBase..+15, K = 32*ks..+31.
__device__ __forceinline__ v16h load_b_frag(const h16* buf, int colBase, int ks, int lane) {
    const h16* p = buf + (size_t)(colBase + (lane & 15)) * DMC + ks * 32 + ((lane >> 4) << 3);
    v8h lo = *(const v8h*)p;
    v8h hi = *(const v8h*)(p + 16);
    v16h r;
#pragma unroll
    for (int i = 0; i < 8; ++i) { r[i] = lo[i]; r[i + 8] = hi[i]; }
    return r;
}

// Store 16x16 f32 accumulator tile to LDS [col][128] f16 at (colBase, chBase), one b128 DS store.
__device__ __forceinline__ void store_d_frag(h16* buf, int colBase, int chBase, int lane, v8f acc) {
    v8h h;
#pragma unroll
    for (int r = 0; r < 8; ++r) h[r] = (_Float16)acc[r];
    *(v8h*)(buf + (size_t)(colBase + (lane & 15)) * DMC + chBase + ((lane >> 4) << 3)) = h;
}

__global__ void prep_weights(const float* lw23, const float* cw, const float* cwf, h16* wf) {
    int i = blockIdx.x * 256 + threadIdx.x;
    if (i < 98304)        wf[i] = (h16)lw23[i];
    else if (i < 147456)  wf[i] = (h16)cw[i - 98304];
    else if (i < WF_TOTAL) wf[i] = (h16)cwf[i - 147456];
}

__global__ __launch_bounds__(256)
__attribute__((amdgpu_waves_per_eu(4)))
void nnue_trunk_heads(
    const float* __restrict__ x,
    const float* __restrict__ lw1,  const float* __restrict__ lb1,
    const h16*   __restrict__ wf,
    const float* __restrict__ lb23,
    const float* __restrict__ cb,   const float* __restrict__ cs,  const float* __restrict__ ct,
    const float* __restrict__ cbf,  const float* __restrict__ csf, const float* __restrict__ ctf,
    const float* __restrict__ pw1,  const float* __restrict__ pb1,
    const float* __restrict__ pw2,  const float* __restrict__ pb2,
    const float* __restrict__ pq,
    const float* __restrict__ vw,   const float* __restrict__ vb,
    const float* __restrict__ vw4,  const float* __restrict__ vb4,
    float* __restrict__ out_v, float* __restrict__ out_p)
{
    extern __shared__ char smem[];
    h16*   stageA = (h16*)  (smem);            // 144 cols * 128 h = 36864 B (3x3 stage)
    h16*   stageB = (h16*)  (smem + 36864);    //  64 cols * 128 h = 16384 B (2x2 stage)
    h16*   tmpB   = (h16*)  (smem + 53248);    //  16 cols * 128 h =  4096 B (ladder out)
    h16*   tmp2   = (h16*)  (smem + 57344);    //  16 cols * 128 h =  4096 B (C3 out)
    float* xin    = (float*)(smem + 57344);    // overlays tmp2: 16*2*16 f32 = 2048 B
    float* featb  = (float*)(smem + 61440);    // 16*48 f32 = 3072 B (lives whole kernel)
    // heads overlays (used after chunk loop, overlap stageA only):
    float* pkA    = (float*)(smem);            // 16*16 f32
    float* pkB    = (float*)(smem + 1024);     // 16*16 f32
    float* vbufA  = (float*)(smem + 2048);     // 16*32 f32
    float* vbufB  = (float*)(smem + 4096);     // 16*32 f32

    const int tid  = threadIdx.x;
    const int lane = tid & 31;
    const int wv   = tid >> 5;         // wave id 0..7 -> output channel rows [16wv,16wv+16)
    const int bbase = blockIdx.x * 16; // 16 boards per workgroup

    for (int i = tid; i < 16 * 48; i += 256) featb[i] = 0.0f;
    __syncthreads();

    const int Y0[8] = {0,0,0,0,3,3,3,3};
    const int X0[8] = {0,0,3,3,3,3,0,0};
    const int RK[8] = {0,2,1,3,2,0,3,1};

    for (int c = 0; c < 8; ++c) {
        // ---- load rotated 4x4x2 windows for 16 boards: xin[img][ch][4][4]
        for (int v = tid; v < 512; v += 256) {
            int img = v >> 5, rem = v & 31, ch = rem >> 4, pos = rem & 15;
            int i4 = pos >> 2, j4 = pos & 3;
            int k = RK[c], sy, sx;
            if (k == 0)      { sy = i4;     sx = j4;     }
            else if (k == 1) { sy = j4;     sx = 3 - i4; }
            else if (k == 2) { sy = 3 - i4; sx = 3 - j4; }
            else             { sy = 3 - j4; sx = i4;     }
            sy += Y0[c]; sx += X0[c];
            int b = bbase + img;
            xin[v] = x[((size_t)(b * 2 + ch) * 7 + sy) * 7 + sx];
        }
        __syncthreads();

        // ---- stage 1: ladder L1 (VALU, K=6) fused with C1 (WMMA), 9 positions -> stageA
        for (int p = 0; p < 9; ++p) {
            int py = p / 3, px = p % 3;
            {   // L1 -> tmpB (each thread: one col, 8 channels)
                int col = tid & 15, mb = (tid >> 4) << 3;
                OPQ(mb);                       // re-load lw1/lb1 per position
                const float* xi = xin + col * 32;
                float a0 = xi[py * 4 + px],            a1 = xi[16 + py * 4 + px];
                float b0 = xi[(py + 1) * 4 + px],      b1 = xi[16 + (py + 1) * 4 + px];
                float c0 = xi[(py + 1) * 4 + px + 1],  c1 = xi[16 + (py + 1) * 4 + px + 1];
                v8h hv;
#pragma unroll
                for (int r = 0; r < 8; ++r) {
                    int m = mb + r;
                    float s = lb1[m]
                            + lw1[(0 * DMC + m) * 2] * a0 + lw1[(0 * DMC + m) * 2 + 1] * a1
                            + lw1[(1 * DMC + m) * 2] * b0 + lw1[(1 * DMC + m) * 2 + 1] * b1
                            + lw1[(2 * DMC + m) * 2] * c0 + lw1[(2 * DMC + m) * 2 + 1] * c1;
                    hv[r] = (_Float16)mishf(s);
                }
                *(v8h*)(tmpB + ((tid & 15)) * DMC + ((tid >> 4) << 3)) = hv;
            }
            __syncthreads();
            {   // C1: cw[0] @ tmpB -> bn+mish -> stageA[p]
                v8f acc = {};
#pragma unroll
                for (int ks = 0; ks < 4; ++ks)
                    acc = wmma_f16(load_a_frag(wf + CW_OFF, wv * 16, ks, lane),
                                   load_b_frag(tmpB, 0, ks, lane), acc);
                int chb = wv * 16 + ((lane >> 4) << 3);
                OPQ(chb);
                v8f post;
#pragma unroll
                for (int r = 0; r < 8; ++r) {
                    int chn = chb + r;
                    float h = (acc[r] + cb[chn]) * cs[chn] + ct[chn];
                    post[r] = mishf(h);
                }
                store_d_frag(stageA, p * 16, wv * 16, lane, post);
            }
            __syncthreads();
        }

        // ---- stage 2: ladder L2 (3 taps, WMMA) + C2, 4 positions -> stageB
        for (int p = 0; p < 4; ++p) {
            int py = p >> 1, px = p & 1;
            int src[3] = { (py * 3 + px) * 16, ((py + 1) * 3 + px) * 16, ((py + 1) * 3 + px + 1) * 16 };
            {
                v8f acc = {};
#pragma unroll
                for (int tap = 0; tap < 3; ++tap) {
                    const h16* W = wf + LW23_OFF + (size_t)tap * 16384;
#pragma unroll
                    for (int ks = 0; ks < 4; ++ks)
                        acc = wmma_f16(load_a_frag(W, wv * 16, ks, lane),
                                       load_b_frag(stageA, src[tap], ks, lane), acc);
                }
                int chb = wv * 16 + ((lane >> 4) << 3);
                OPQ(chb);
                v8f post;
#pragma unroll
                for (int r = 0; r < 8; ++r)
                    post[r] = mishf(acc[r] + lb23[chb + r]);
                store_d_frag(tmpB, 0, wv * 16, lane, post);
            }
            __syncthreads();
            {   // C2: cw[1] @ tmpB -> bn+mish -> stageB[p]
                v8f acc = {};
#pragma unroll
                for (int ks = 0; ks < 4; ++ks)
                    acc = wmma_f16(load_a_frag(wf + CW_OFF + 16384, wv * 16, ks, lane),
                                   load_b_frag(tmpB, 0, ks, lane), acc);
                int chb = wv * 16 + ((lane >> 4) << 3);
                OPQ(chb);
                v8f post;
#pragma unroll
                for (int r = 0; r < 8; ++r) {
                    int chn = chb + r;
                    float h = (acc[r] + cb[DMC + chn]) * cs[DMC + chn] + ct[DMC + chn];
                    post[r] = mishf(h);
                }
                store_d_frag(stageB, p * 16, wv * 16, lane, post);
            }
            __syncthreads();
        }

        // ---- stage 3: ladder L3 -> C3 -> CF
        {
            int src[3] = { 0, 2 * 16, 3 * 16 };   // taps (0,0),(1,0),(1,1) in 2x2 grid
            v8f acc = {};
#pragma unroll
            for (int tap = 0; tap < 3; ++tap) {
                const h16* W = wf + LW23_OFF + (size_t)(3 + tap) * 16384;
#pragma unroll
                for (int ks = 0; ks < 4; ++ks)
                    acc = wmma_f16(load_a_frag(W, wv * 16, ks, lane),
                                   load_b_frag(stageB, src[tap], ks, lane), acc);
            }
            int chb = wv * 16 + ((lane >> 4) << 3);
            OPQ(chb);
            v8f post;
#pragma unroll
            for (int r = 0; r < 8; ++r)
                post[r] = mishf(acc[r] + lb23[DMC + chb + r]);
            store_d_frag(tmpB, 0, wv * 16, lane, post);
        }
        __syncthreads();
        {   // C3: cw[2] @ tmpB -> bn+mish -> tmp2
            v8f acc = {};
#pragma unroll
            for (int ks = 0; ks < 4; ++ks)
                acc = wmma_f16(load_a_frag(wf + CW_OFF + 2 * 16384, wv * 16, ks, lane),
                               load_b_frag(tmpB, 0, ks, lane), acc);
            int chb = wv * 16 + ((lane >> 4) << 3);
            OPQ(chb);
            v8f post;
#pragma unroll
            for (int r = 0; r < 8; ++r) {
                int chn = chb + r;
                float h = (acc[r] + cb[2 * DMC + chn]) * cs[2 * DMC + chn] + ct[2 * DMC + chn];
                post[r] = mishf(h);
            }
            store_d_frag(tmp2, 0, wv * 16, lane, post);
        }
        __syncthreads();
        // CF: cwf (48x128) @ tmp2, quantize, accumulate feat (waves 0..2; unique (img,ch) per lane)
        if (wv < 3) {
            v8f acc = {};
#pragma unroll
            for (int ks = 0; ks < 4; ++ks)
                acc = wmma_f16(load_a_frag(wf + CWF_OFF, wv * 16, ks, lane),
                               load_b_frag(tmp2, 0, ks, lane), acc);
            int img = lane & 15;
            int chb = wv * 16 + ((lane >> 4) << 3);
            OPQ(chb);
#pragma unroll
            for (int r = 0; r < 8; ++r) {
                int chn = chb + r;
                float h = (acc[r] + cbf[chn]) * csf[chn] + ctf[chn];
                float q = fminf(fmaxf(h, -1.0f), QMAXF);
                q = rintf(q * 128.0f) * (1.0f / 128.0f);
                featb[img * 48 + chn] += q;
            }
        }
        __syncthreads();   // protects tmp2/xin overlay + featb before next chunk
    }

    // ================= fused heads (feat complete for these 16 boards) =================
    {   // policy L1: 16 img x 16 neurons, one per thread
        int img = tid >> 4, o = tid & 15;
        float s = pb1[o];
#pragma unroll
        for (int k = 0; k < 16; ++k) s += featb[img * 48 + k] * pw1[o * 16 + k];
        pkA[img * 16 + o] = fmaxf(s, 0.0f);
    }
    for (int v = tid; v < 512; v += 256) {   // value init: v = feat[:,16:48]
        int img = v >> 5, k = v & 31;
        vbufA[v] = featb[img * 48 + 16 + k];
    }
    __syncthreads();
    {   // policy L2
        int img = tid >> 4, o = tid & 15;
        float s = pb2[o];
#pragma unroll
        for (int k = 0; k < 16; ++k) s += pkA[img * 16 + k] * pw2[o * 16 + k];
        pkB[img * 16 + o] = s;
    }
    // value trunk: 3 layers of 32x32 with fake-quantized weights
    float* srcv = vbufA; float* dstv = vbufB;
    for (int L = 0; L < 3; ++L) {
        __syncthreads();
        int img = tid >> 4, ob = (tid & 15) << 1;
#pragma unroll
        for (int oo = 0; oo < 2; ++oo) {
            int o = ob + oo;
            float s = rintf(vb[L * 32 + o] * 16384.0f) * (1.0f / 16384.0f);
            for (int k = 0; k < 32; ++k) {
                float w  = vw[((size_t)L * 32 + o) * 32 + k];
                float qw = rintf(fminf(fmaxf(w, -1.0f), QMAXF) * 128.0f) * (1.0f / 128.0f);
                float vv = fminf(fmaxf(srcv[img * 32 + k], 0.0f), QMAXF);
                s += vv * qw;
            }
            dstv[img * 32 + o] = s;
        }
        float* t = srcv; srcv = dstv; dstv = t;
    }
    __syncthreads();
    if (tid < 48) {   // final value layer: 16 img x 3
        int img = tid / 3, o = tid % 3;
        float s = rintf(vb4[o] * 16384.0f) * (1.0f / 16384.0f);
        for (int k = 0; k < 32; ++k) {
            float qw = rintf(fminf(fmaxf(vw4[o * 32 + k], -1.0f), QMAXF) * 128.0f) * (1.0f / 128.0f);
            float vv = fminf(fmaxf(srcv[img * 32 + k], 0.0f), QMAXF);
            s += vv * qw;
        }
        out_v[(size_t)(bbase + img) * 3 + o] = s;
    }
    for (int idx = tid; idx < 16 * 49; idx += 256) {   // policy projection 16 -> 49
        int img = idx / 49, o = idx % 49;
        float s = 0.0f;
#pragma unroll
        for (int k = 0; k < 16; ++k) s += pkB[img * 16 + k] * pq[o * 16 + k];
        out_p[(size_t)(bbase + img) * 49 + o] = s;
    }
}

extern "C" void kernel_launch(void* const* d_in, const int* in_sizes, int n_in,
                              void* d_out, int out_size, void* d_ws, size_t ws_size,
                              hipStream_t stream) {
    const float* x    = (const float*)d_in[0];
    const float* lw1  = (const float*)d_in[1];
    const float* lb1  = (const float*)d_in[2];
    const float* lw23 = (const float*)d_in[3];
    const float* lb23 = (const float*)d_in[4];
    const float* cw   = (const float*)d_in[5];
    const float* cb   = (const float*)d_in[6];
    const float* cs   = (const float*)d_in[7];
    const float* ct   = (const float*)d_in[8];
    const float* cwf  = (const float*)d_in[9];
    const float* cbf  = (const float*)d_in[10];
    const float* csf  = (const float*)d_in[11];
    const float* ctf  = (const float*)d_in[12];
    const float* pw1  = (const float*)d_in[13];
    const float* pb1  = (const float*)d_in[14];
    const float* pw2  = (const float*)d_in[15];
    const float* pb2  = (const float*)d_in[16];
    const float* pq   = (const float*)d_in[17];
    const float* vw   = (const float*)d_in[18];
    const float* vb   = (const float*)d_in[19];
    const float* vw4  = (const float*)d_in[20];
    const float* vb4  = (const float*)d_in[21];

    h16*   wf  = (h16*)d_ws;
    float* out = (float*)d_out;

    prep_weights<<<(WF_TOTAL + 255) / 256, 256, 0, stream>>>(lw23, cw, cwf, wf);

    nnue_trunk_heads<<<NBAT / 16, 256, 64512, stream>>>(
        x, lw1, lb1, wf, lb23, cb, cs, ct, cbf, csf, ctf,
        pw1, pb1, pw2, pb2, pq, vw, vb, vw4, vb4,
        out /* v: B*3 */, out + (size_t)NBAT * 3 /* policy: B*49 */);
}
